// DualMemoryMamba_17557826306472
// MI455X (gfx1250) — matrix operand order repaired and checked by reading, not compiled, not run on hardware
//
#include <hip/hip_runtime.h>
#include <hip/hip_bf16.h>

// ---------------- problem constants (match reference) ----------------
constexpr int BATCH  = 2;
constexpr int SEQ    = 2048;
constexpr int DIM    = 1024;
constexpr int DSTATE = 16;
constexpr int DINNER = 2048;                 // EXPAND * DIM
constexpr int DTRANK = 64;                   // DIM / 16
constexpr int NX     = DTRANK + 2 * DSTATE;  // 96 (valid dbc cols)
constexpr int NXP    = 128;                  // padded dbc width (GEMM-friendly)
constexpr int MROWS  = BATCH * SEQ;          // 4096 GEMM rows

// ---------------- WMMA types ----------------
typedef __attribute__((ext_vector_type(16))) __bf16 v16bf;
typedef __attribute__((ext_vector_type(8)))  float  v8f;

struct __attribute__((aligned(16))) U4 { unsigned x, y, z, w; };
union FragBF { U4 u[2]; v16bf v; };

__device__ __forceinline__ unsigned short f2bf(float x) {
  unsigned u = __float_as_uint(x);
  unsigned r = ((u >> 16) & 1u) + 0x7FFFu;   // round-to-nearest-even
  return (unsigned short)((u + r) >> 16);
}
__device__ __forceinline__ float fsilu(float x) { return x / (1.f + __expf(-x)); }

// =====================================================================
// bf16 GEMM: D[M,N] = A[M,K] * W[N,K]^T,  M,N multiples of 128, K of 32.
// 256 threads = 8 waves as 4(M) x 2(N); wave tile 32x64 -> block 128x128.
// A/B 128x32 tiles double-buffered in LDS, filled with
// global_load_async_to_lds_b128 (ASYNCcnt) while WMMAs consume prev stage.
// LDS tile layout: row-major, 64 bytes (32 bf16) per row.
// =====================================================================
__global__ __launch_bounds__(256) void k_gemm_bf16(
    const unsigned short* __restrict__ A, const unsigned short* __restrict__ W,
    float* __restrict__ D, int K, int lda, int ldw, int ldd) {
  __shared__ U4 smA[2][512];   // 2 x 8KB
  __shared__ U4 smB[2][512];

  const int tid  = threadIdx.x;
  const int lane = tid & 31;
  const int wave = tid >> 5;
  const int mw = wave & 3, nw = wave >> 2;
  const int mBase = blockIdx.y * 128;
  const int nBase = blockIdx.x * 128;

  // per-thread async-copy job: 32 bytes of the A tile + 32 bytes of the B tile
  const int crow  = tid >> 1;          // tile row 0..127
  const int chalf = tid & 1;           // which 32B half of the 64B row
  const unsigned short* gA = A + (size_t)(mBase + crow) * lda + chalf * 16;
  const unsigned short* gB = W + (size_t)(nBase + crow) * ldw + chalf * 16;
  const unsigned coff = (unsigned)(crow * 64 + chalf * 32);
  unsigned ldsA[2], ldsB[2];
  ldsA[0] = (unsigned)(size_t)(&smA[0][0]) + coff;
  ldsA[1] = (unsigned)(size_t)(&smA[1][0]) + coff;
  ldsB[0] = (unsigned)(size_t)(&smB[0][0]) + coff;
  ldsB[1] = (unsigned)(size_t)(&smB[1][0]) + coff;

  auto stage = [&](int buf, int k0) {
    const unsigned short* pa = gA + k0;
    const unsigned short* pb = gB + k0;
    asm volatile("global_load_async_to_lds_b128 %0, %1, off"
                 :: "v"(ldsA[buf]), "v"(pa) : "memory");
    asm volatile("global_load_async_to_lds_b128 %0, %1, off offset:16"
                 :: "v"(ldsA[buf]), "v"(pa) : "memory");
    asm volatile("global_load_async_to_lds_b128 %0, %1, off"
                 :: "v"(ldsB[buf]), "v"(pb) : "memory");
    asm volatile("global_load_async_to_lds_b128 %0, %1, off offset:16"
                 :: "v"(ldsB[buf]), "v"(pb) : "memory");
  };

  v8f acc[2][4] = {};
  const int nsteps = K >> 5;
  stage(0, 0);
  for (int s = 0; s < nsteps; ++s) {
    const int buf = s & 1;
    if (s + 1 < nsteps) {
      stage(buf ^ 1, (s + 1) << 5);
      asm volatile("s_wait_asynccnt 0x4" ::: "memory");  // prev stage landed
    } else {
      asm volatile("s_wait_asynccnt 0x0" ::: "memory");
    }
    __syncthreads();

    // A fragment: lane m=lane&15; K {0..7,16..23} (lanes 0-15) / {8..15,24..31}
    FragBF af[2];
#pragma unroll
    for (int mt = 0; mt < 2; ++mt) {
      int r = mw * 32 + mt * 16 + (lane & 15);
      int c = (lane >> 4);                      // U4 (16B) granules
      af[mt].u[0] = smA[buf][r * 4 + c];
      af[mt].u[1] = smA[buf][r * 4 + c + 2];
    }
    // B fragment: lane n=lane&15; K {0..7,16..23} (lanes 0-15) / {8..15,24..31}
    FragBF bfr[4];
#pragma unroll
    for (int nt = 0; nt < 4; ++nt) {
      int r = nw * 64 + nt * 16 + (lane & 15);
      int c = (lane >> 4);
      bfr[nt].u[0] = smB[buf][r * 4 + c];
      bfr[nt].u[1] = smB[buf][r * 4 + c + 2];
    }
#pragma unroll
    for (int mt = 0; mt < 2; ++mt)
#pragma unroll
      for (int nt = 0; nt < 4; ++nt)
        acc[mt][nt] = __builtin_amdgcn_wmma_f32_16x16x32_bf16(
            false, af[mt].v, false, bfr[nt].v, (short)0, acc[mt][nt], false, false);
    __syncthreads();   // all reads done before this buf is overwritten
  }

  // C/D layout: VGPR v -> M = v (lanes 0-15) or v+8 (lanes 16-31); lane&15 = N
  const int rOff = (lane >> 4) << 3;
#pragma unroll
  for (int mt = 0; mt < 2; ++mt) {
    const int mrow = mBase + mw * 32 + mt * 16 + rOff;
#pragma unroll
    for (int nt = 0; nt < 4; ++nt) {
      const int col = nBase + nw * 64 + nt * 16 + (lane & 15);
#pragma unroll
      for (int v = 0; v < 8; ++v)
        D[(size_t)(mrow + v) * ldd + col] = acc[mt][nt][v];
    }
  }
}

// ---------------- elementwise kernels ----------------
__global__ __launch_bounds__(256) void k_f32_to_bf16(
    const float* __restrict__ s, unsigned short* __restrict__ d, int n) {
  int i = blockIdx.x * 256 + threadIdx.x;
  if (i < n) d[i] = f2bf(s[i]);
}

// convert with zero row padding (for xproj: 96 real rows -> 128)
__global__ __launch_bounds__(256) void k_cvt_pad(
    const float* __restrict__ s, unsigned short* __restrict__ d,
    int rowsSrc, int rowsDst, int cols) {
  int i = blockIdx.x * 256 + threadIdx.x;
  if (i >= rowsDst * cols) return;
  int r = i / cols;
  d[i] = (r < rowsSrc) ? f2bf(s[i]) : (unsigned short)0;
}

// x_masked = h*m + (m*mpw + mpb)*(1-m); also emit bf16 copy for GEMM1
__global__ __launch_bounds__(256) void k_mask_blend(
    const float* __restrict__ hidden, const float* __restrict__ mask,
    const float* __restrict__ mpw, const float* __restrict__ mpb,
    float* __restrict__ xm, unsigned short* __restrict__ xm_bf, int total) {
  int i = blockIdx.x * 256 + threadIdx.x;
  if (i >= total) return;
  int d  = i % DIM;
  int bl = i / DIM;
  float m  = mask[bl];
  float mf = m * mpw[d] + mpb[d];
  float x  = hidden[i] * m + mf * (1.f - m);
  xm[i]    = x;
  xm_bf[i] = f2bf(x);
}

// depthwise causal conv over time + bias + SiLU. dir=0: taps x[l-3+j]; dir=1: x[l+3-j]
__global__ __launch_bounds__(256) void k_conv_silu(
    const float* __restrict__ xz, const float* __restrict__ cw,
    const float* __restrict__ cb, float* __restrict__ u,
    unsigned short* __restrict__ u_bf, int dir) {
  int i = blockIdx.x * 256 + threadIdx.x;
  if (i >= BATCH * SEQ * DINNER) return;
  int d  = i % DINNER;
  int bl = i / DINNER;
  int l  = bl % SEQ;
  int b  = bl / SEQ;
  const float* base = xz + (size_t)b * SEQ * (2 * DINNER) + d;  // xi channel d
  float acc = cb[d];
#pragma unroll
  for (int j = 0; j < 4; ++j) {
    int p = (dir == 0) ? (l - 3 + j) : (l + 3 - j);
    if (p >= 0 && p < SEQ) acc += cw[d * 4 + j] * base[(size_t)p * (2 * DINNER)];
  }
  float s = fsilu(acc);
  u[i]    = s;
  u_bf[i] = f2bf(s);
}

// slice dt = dbc[:, 0:64] -> packed bf16 for the delta GEMM
__global__ __launch_bounds__(256) void k_extract_dt(
    const float* __restrict__ dbc, unsigned short* __restrict__ dt_bf, int rows) {
  int i = blockIdx.x * 256 + threadIdx.x;
  if (i >= rows * DTRANK) return;
  int r = i >> 6, c = i & 63;
  dt_bf[i] = f2bf(dbc[(size_t)r * NXP + c]);
}

// fused: softplus(delta_pre + dt_b) -> selective scan -> (+u*D)*silu(z) -> bf16 y
// one thread per (batch, channel); 16 SSM states live in registers.
__global__ __launch_bounds__(256) void k_scan(
    const float* __restrict__ dpre, const float* __restrict__ dt_b,
    const float* __restrict__ u, const float* __restrict__ dbc,
    const float* __restrict__ xz, const float* __restrict__ A_log,
    const float* __restrict__ Dvec, unsigned short* __restrict__ y_bf, int dir) {
  int d = blockIdx.x * 256 + threadIdx.x;
  int b = blockIdx.y;
  if (d >= DINNER) return;
  float An[DSTATE], h[DSTATE];
#pragma unroll
  for (int n = 0; n < DSTATE; ++n) {
    An[n] = -__expf(A_log[d * DSTATE + n]);   // A = -exp(A_log)
    h[n]  = 0.f;
  }
  const float Dd  = Dvec[d];
  const float bia = dt_b[d];
  for (int s = 0; s < SEQ; ++s) {
    int t = (dir == 0) ? s : (SEQ - 1 - s);
    size_t row = (size_t)b * SEQ + t;
    float dp    = dpre[row * DINNER + d] + bia;
    float delta = (dp > 20.f) ? dp : log1pf(__expf(dp));
    float ut    = u[row * DINNER + d];
    float du    = delta * ut;
    const float* bc = dbc + row * NXP;        // [dt(64) | B(16) | C(16) | pad]
    float y = 0.f;
#pragma unroll
    for (int n = 0; n < DSTATE; ++n) {
      float dA = __expf(delta * An[n]);
      h[n] = dA * h[n] + du * bc[DTRANK + n];
      y += h[n] * bc[DTRANK + DSTATE + n];
    }
    float z = xz[row * (2 * DINNER) + DINNER + d];
    y_bf[row * DINNER + d] = f2bf((y + ut * Dd) * fsilu(z));
  }
}

// out = (f_out + b_out)*m + hidden*(1-m)
__global__ __launch_bounds__(256) void k_final(
    const float* __restrict__ fo, const float* __restrict__ bo,
    const float* __restrict__ hidden, const float* __restrict__ mask,
    float* __restrict__ out, int total) {
  int i = blockIdx.x * 256 + threadIdx.x;
  if (i >= total) return;
  float m = mask[i / DIM];
  out[i] = (fo[i] + bo[i]) * m + hidden[i] * (1.f - m);
}

// ---------------- host orchestration ----------------
extern "C" void kernel_launch(void* const* d_in, const int* in_sizes, int n_in,
                              void* d_out, int out_size, void* d_ws, size_t ws_size,
                              hipStream_t stream) {
  (void)in_sizes; (void)n_in; (void)out_size; (void)ws_size;
  const float* hidden = (const float*)d_in[0];
  const float* mask   = (const float*)d_in[1];
  const float* mpw    = (const float*)d_in[2];
  const float* mpb    = (const float*)d_in[3];
  // per-direction weights: in_w, conv_w, conv_b, xproj_w, dt_w, dt_b, A_log, D, out_w
  auto Wp = [&](int dir, int k) { return (const float*)d_in[4 + dir * 9 + k]; };

  char* ws = (char*)d_ws;
  size_t off = 0;
  auto alloc = [&](size_t bytes) -> void* {
    off = (off + 255) & ~(size_t)255;
    void* p = ws + off;
    off += bytes;
    return p;
  };

  float*          xm     = (float*)alloc((size_t)MROWS * DIM * 4);
  unsigned short* xm_bf  = (unsigned short*)alloc((size_t)MROWS * DIM * 2);
  unsigned short* inw_bf[2], *xpj_bf[2], *dtw_bf[2], *ow_bf[2];
  for (int d2 = 0; d2 < 2; ++d2) {
    inw_bf[d2] = (unsigned short*)alloc((size_t)2 * DINNER * DIM * 2);
    xpj_bf[d2] = (unsigned short*)alloc((size_t)NXP * DINNER * 2);   // 128 rows (32 zero)
    dtw_bf[d2] = (unsigned short*)alloc((size_t)DINNER * DTRANK * 2);
    ow_bf[d2]  = (unsigned short*)alloc((size_t)DIM * DINNER * 2);
  }
  float*          xz    = (float*)alloc((size_t)MROWS * 2 * DINNER * 4);
  float*          u     = (float*)alloc((size_t)MROWS * DINNER * 4);
  unsigned short* u_bf  = (unsigned short*)alloc((size_t)MROWS * DINNER * 2);
  float*          dbc   = (float*)alloc((size_t)MROWS * NXP * 4);
  unsigned short* dt_bf = (unsigned short*)alloc((size_t)MROWS * DTRANK * 2);
  float*          dpre  = (float*)alloc((size_t)MROWS * DINNER * 4);
  unsigned short* y_bf  = (unsigned short*)alloc((size_t)MROWS * DINNER * 2);
  float*          dout[2];
  dout[0] = (float*)alloc((size_t)MROWS * DIM * 4);
  dout[1] = (float*)alloc((size_t)MROWS * DIM * 4);

  const int totalX = MROWS * DIM;
  k_mask_blend<<<(totalX + 255) / 256, 256, 0, stream>>>(
      hidden, mask, mpw, mpb, xm, xm_bf, totalX);

  auto cvt = [&](const float* s, unsigned short* d2, int n) {
    k_f32_to_bf16<<<(n + 255) / 256, 256, 0, stream>>>(s, d2, n);
  };
  for (int d2 = 0; d2 < 2; ++d2) {
    cvt(Wp(d2, 0), inw_bf[d2], 2 * DINNER * DIM);
    k_cvt_pad<<<(NXP * DINNER + 255) / 256, 256, 0, stream>>>(
        Wp(d2, 3), xpj_bf[d2], NX, NXP, DINNER);
    cvt(Wp(d2, 4), dtw_bf[d2], DINNER * DTRANK);
    cvt(Wp(d2, 8), ow_bf[d2], DIM * DINNER);
  }

  const dim3 blk(256);
  const int mT = MROWS / 128;  // 32
  for (int dir = 0; dir < 2; ++dir) {
    // xz = x_masked @ in_w^T   [4096 x 4096 x 1024]
    k_gemm_bf16<<<dim3((2 * DINNER) / 128, mT), blk, 0, stream>>>(
        xm_bf, inw_bf[dir], xz, DIM, DIM, DIM, 2 * DINNER);
    // u = silu(conv(xi) + b)
    k_conv_silu<<<(MROWS * DINNER + 255) / 256, blk, 0, stream>>>(
        xz, Wp(dir, 1), Wp(dir, 2), u, u_bf, dir);
    // dbc = u @ xproj_pad^T   [4096 x 128 x 2048]
    k_gemm_bf16<<<dim3(NXP / 128, mT), blk, 0, stream>>>(
        u_bf, xpj_bf[dir], dbc, DINNER, DINNER, DINNER, NXP);
    // dt slice -> bf16
    k_extract_dt<<<(MROWS * DTRANK + 255) / 256, blk, 0, stream>>>(dbc, dt_bf, MROWS);
    // delta_pre = dt @ dt_w^T   [4096 x 2048 x 64]
    k_gemm_bf16<<<dim3(DINNER / 128, mT), blk, 0, stream>>>(
        dt_bf, dtw_bf[dir], dpre, DTRANK, DTRANK, DTRANK, DINNER);
    // fused softplus + selective scan + skip + gate -> y_bf
    k_scan<<<dim3(DINNER / 256, BATCH), blk, 0, stream>>>(
        dpre, Wp(dir, 5), u, dbc, xz, Wp(dir, 6), Wp(dir, 7), y_bf, dir);
    // dir_out = y @ out_w^T   [4096 x 1024 x 2048]
    k_gemm_bf16<<<dim3(DIM / 128, mT), blk, 0, stream>>>(
        y_bf, ow_bf[dir], dout[dir], DINNER, DINNER, DINNER, DIM);
  }

  k_final<<<(totalX + 255) / 256, blk, 0, stream>>>(
      dout[0], dout[1], hidden, mask, (float*)d_out, totalX);
}
